// BiLSTM_CRF_81638738362762
// MI455X (gfx1250) — compile-verified, matrix operand order checked by reading, not run on hardware
//
#include <hip/hip_runtime.h>
#include <hip/hip_bf16.h>

#define BB 64
#define TTT 512
#define EE 128
#define HH 256
#define KK 20

typedef __attribute__((ext_vector_type(16))) __bf16 v16bf;
typedef __attribute__((ext_vector_type(8)))  __bf16 v8bf;
typedef __attribute__((ext_vector_type(8)))  float  v8f;

static __device__ __forceinline__ float sigm(float x) { return 1.0f / (1.0f + __expf(-x)); }

// ---- WMMA A-operand load from row-major bf16 global memory -----------------
static __device__ __forceinline__ v16bf load_a16(const __bf16* A, int lda, int m0, int k0) {
  int lane = threadIdx.x & 31;
  int row  = m0 + (lane & 15);
  int kb   = k0 + ((lane < 16) ? 0 : 8);
  const __bf16* p = A + (size_t)row * lda + kb;
  v8bf lo = *(const v8bf*)p;          // K kb .. kb+7
  v8bf hi = *(const v8bf*)(p + 16);   // K kb+16 .. kb+23
  v16bf r;
#pragma unroll
  for (int i = 0; i < 8; ++i) { r[i] = lo[i]; r[i + 8] = hi[i]; }
  return r;
}

// ---- WMMA B-operand load from pre-packed tiles (32B contiguous per lane) ---
static __device__ __forceinline__ v16bf load_bp(const __bf16* Bp, int ntiles, int kt, int nt) {
  int lane = threadIdx.x & 31;
  const __bf16* p = Bp + (((size_t)(kt * ntiles + nt)) << 9) + (lane << 4);
  return *(const v16bf*)p;
}

// ---- embedding gather + f32 -> bf16, rows permuted to (t*B + b) ------------
__global__ void embed_gather(const int* __restrict__ x, const float* __restrict__ emb,
                             __bf16* __restrict__ out, long total) {
  long i = (long)blockIdx.x * blockDim.x + threadIdx.x;
  long stride = (long)gridDim.x * blockDim.x;
  for (; i < total; i += stride) {
    long r   = i >> 7;              // permuted row = t*64 + b
    long col = i & 127;
    long t = r >> 6, b = r & 63;
    out[i] = (__bf16)emb[((size_t)x[b * TTT + t] << 7) + col];
  }
}

// ---- pack W (Ndim x Kdim row-major, B = W^T) into WMMA B-operand tiles -----
__global__ void pack_b(const float* __restrict__ W, __bf16* __restrict__ out,
                       int Kdim, int Ndim, int Nvalid) {
  size_t total = (size_t)Kdim * Ndim;
  size_t i = (size_t)blockIdx.x * blockDim.x + threadIdx.x;
  size_t stride = (size_t)gridDim.x * blockDim.x;
  int ntiles = Ndim >> 4;
  for (; i < total; i += stride) {
    int idx  = (int)(i & 15);
    int lane = (int)((i >> 4) & 31);
    size_t tile = i >> 9;
    int nt = (int)(tile % ntiles);
    int kt = (int)(tile / ntiles);
    int kk = (idx & 7) + ((idx >= 8) ? 16 : 0) + ((lane >= 16) ? 8 : 0);
    int n  = nt * 16 + (lane & 15);
    int k  = kt * 32 + kk;
    float v = (n < Nvalid) ? W[(size_t)n * Kdim + k] : 0.0f;
    out[i] = (__bf16)v;
  }
}

// ---- generic WMMA GEMM ------------------------------------------------------
// OUT_MODE 0: f32 row-major (ldout), store n < Nvalid only
// OUT_MODE 1: bf16 packed C-tile layout: [(mt*Ntiles+nt)*32 + lane]*8 + v
template <int OUT_MODE>
__global__ void gemm_bpacked(const __bf16* __restrict__ A, const __bf16* __restrict__ Bp,
                             const float* __restrict__ bias, void* __restrict__ Out,
                             int M, int Kdim, int Ntiles, int Nvalid, int ldout) {
  int wave = blockIdx.x * (blockDim.x >> 5) + (threadIdx.x >> 5);
  int mtiles = M >> 4;
  if (wave >= mtiles * Ntiles) return;
  int mt = wave / Ntiles, nt = wave % Ntiles;
  int lane = threadIdx.x & 31;
  int n = nt * 16 + (lane & 15);
  float bv = (n < Nvalid) ? bias[n] : 0.0f;
  v8f acc;
#pragma unroll
  for (int i = 0; i < 8; ++i) acc[i] = bv;
  int kts = Kdim >> 5;
  for (int kt = 0; kt < kts; ++kt) {
    v16bf a = load_a16(A, Kdim, mt << 4, kt << 5);
    v16bf b = load_bp(Bp, Ntiles, kt, nt);
    acc = __builtin_amdgcn_wmma_f32_16x16x32_bf16(false, a, false, b, (short)0, acc, false, false);
  }
  if (OUT_MODE == 1) {
    v8bf ov;
#pragma unroll
    for (int v = 0; v < 8; ++v) ov[v] = (__bf16)acc[v];
    *(v8bf*)((__bf16*)Out + (((size_t)(mt * Ntiles + nt) * 32 + lane) << 3)) = ov;
  } else if (n < Nvalid) {
#pragma unroll
    for (int v = 0; v < 8; ++v) {
      int m = (mt << 4) + v + ((lane >= 16) ? 8 : 0);
      ((float*)Out)[(size_t)m * ldout + n] = acc[v];
    }
  }
}

// ---- persistent bidirectional LSTM: grid.x = direction ---------------------
// xg is in packed C-tile layout with permuted rows (t*B+b): row tile = t*4+mt.
__global__ void __launch_bounds__(1024) lstm_kernel(
    const __bf16* __restrict__ xg_f, const __bf16* __restrict__ xg_b,
    const __bf16* __restrict__ whhp_f, const __bf16* __restrict__ whhp_b,
    __bf16* __restrict__ feats) {
  const int dir = blockIdx.x;
  const __bf16* xg = dir ? xg_b : xg_f;
  unsigned long long wpi = (unsigned long long)(dir ? whhp_b : whhp_f);

  // h in WMMA A-operand layout, double buffered: [buf][mt(4)][kt(8)][lane(32)][16]
  __shared__ __align__(32) __bf16 hbuf[2][4][8][32][16];   // 64 KB

  const int tid  = threadIdx.x;
  const int wave = tid >> 5, lane = tid & 31;
  const int j    = wave >> 1;          // h-column tile 0..15
  const int mt0  = (wave & 1) * 2;     // batch row-tile pair
  const int nl   = lane & 15;
  const int mb   = (lane >= 16) ? 8 : 0;

  for (int i = tid; i < 2 * 4 * 8 * 32 * 16; i += blockDim.x) ((__bf16*)hbuf)[i] = (__bf16)0.0f;
  __syncthreads();

  v8f c[2];
#pragma unroll
  for (int r = 0; r < 2; ++r)
#pragma unroll
    for (int v = 0; v < 8; ++v) c[r][v] = 0.0f;

  for (int t = 0; t < TTT; ++t) {
    const int ttm = dir ? (TTT - 1 - t) : t;
    const int cur = t & 1, nxt = cur ^ 1;

    // Defeat loop-invariant hoisting of the (t-invariant) weight tiles: they
    // must stream from L2 each step, not live in (spilled) registers.
    asm volatile("" : "+s"(wpi));
    const __bf16* wp = (const __bf16*)wpi;

    // init gate accumulators from packed xg: one 16B vector load per tile
    v8f ai[2], af[2], ag[2], ao[2];
#pragma unroll
    for (int r = 0; r < 2; ++r) {
      const size_t rowt = (size_t)(ttm * 4 + mt0 + r) * 64;
      v8bf ti = *(const v8bf*)(xg + (((rowt + j) * 32 + lane) << 3));
      v8bf tf = *(const v8bf*)(xg + (((rowt + 16 + j) * 32 + lane) << 3));
      v8bf tg = *(const v8bf*)(xg + (((rowt + 32 + j) * 32 + lane) << 3));
      v8bf to = *(const v8bf*)(xg + (((rowt + 48 + j) * 32 + lane) << 3));
#pragma unroll
      for (int v = 0; v < 8; ++v) {
        ai[r][v] = (float)ti[v];
        af[r][v] = (float)tf[v];
        ag[r][v] = (float)tg[v];
        ao[r][v] = (float)to[v];
      }
    }

    // gates += h_{t-1} @ Whh^T  (K=256 -> 8 k-steps); one B tile live at a time
#pragma unroll
    for (int kt = 0; kt < 8; ++kt) {
      const v16bf a0 = *(const v16bf*)&hbuf[cur][mt0][kt][lane][0];
      const v16bf a1 = *(const v16bf*)&hbuf[cur][mt0 + 1][kt][lane][0];
      {
        const v16bf b = load_bp(wp, 64, kt, j);
        ai[0] = __builtin_amdgcn_wmma_f32_16x16x32_bf16(false, a0, false, b, (short)0, ai[0], false, false);
        ai[1] = __builtin_amdgcn_wmma_f32_16x16x32_bf16(false, a1, false, b, (short)0, ai[1], false, false);
      }
      {
        const v16bf b = load_bp(wp, 64, kt, 16 + j);
        af[0] = __builtin_amdgcn_wmma_f32_16x16x32_bf16(false, a0, false, b, (short)0, af[0], false, false);
        af[1] = __builtin_amdgcn_wmma_f32_16x16x32_bf16(false, a1, false, b, (short)0, af[1], false, false);
      }
      {
        const v16bf b = load_bp(wp, 64, kt, 32 + j);
        ag[0] = __builtin_amdgcn_wmma_f32_16x16x32_bf16(false, a0, false, b, (short)0, ag[0], false, false);
        ag[1] = __builtin_amdgcn_wmma_f32_16x16x32_bf16(false, a1, false, b, (short)0, ag[1], false, false);
      }
      {
        const v16bf b = load_bp(wp, 64, kt, 48 + j);
        ao[0] = __builtin_amdgcn_wmma_f32_16x16x32_bf16(false, a0, false, b, (short)0, ao[0], false, false);
        ao[1] = __builtin_amdgcn_wmma_f32_16x16x32_bf16(false, a1, false, b, (short)0, ao[1], false, false);
      }
    }

    // cell update (i/f/g/o for a given (m, hcol) are all in this lane)
#pragma unroll
    for (int r = 0; r < 2; ++r) {
#pragma unroll
      for (int v = 0; v < 8; ++v) {
        float iv = sigm(ai[r][v]);
        float fv = sigm(af[r][v]);
        float gv = tanhf(ag[r][v]);
        float ov = sigm(ao[r][v]);
        float cc = fv * c[r][v] + iv * gv;
        c[r][v] = cc;
        float h = ov * tanhf(cc);

        int m = ((mt0 + r) << 4) + v + mb;   // batch
        int k = (j << 4) + nl;               // hidden col
        // feats row-major, permuted rows (t*B + b)
        feats[(((size_t)(ttm * 64 + m)) << 9) + (dir << 8) + k] = (__bf16)h;

        // scatter into next step's A-operand layout in LDS
        int kk  = k & 31;
        int hi  = (kk >> 3) & 1;
        int idx = (kk & 7) + ((kk >= 16) ? 8 : 0);
        hbuf[nxt][mt0 + r][k >> 5][(m & 15) + (hi << 4)][idx] = (__bf16)h;
      }
    }
    __syncthreads();
  }
}

// ---- Viterbi: one wave per batch element (emissions rows are t*B + b) ------
__global__ void viterbi_kernel(const float* __restrict__ emis,
                               const unsigned char* __restrict__ mask,
                               const float* __restrict__ start_t,
                               const float* __restrict__ end_t,
                               const float* __restrict__ trans,
                               int* __restrict__ out) {
  __shared__ int hist[(TTT - 1) * KK];   // ~40 KB
  const int b = blockIdx.x;
  const int j = threadIdx.x;
  const bool act = j < KK;

  float tc[KK];
#pragma unroll
  for (int i = 0; i < KK; ++i) tc[i] = act ? trans[i * KK + j] : 0.0f;

  float score = act ? (start_t[j] + emis[(size_t)b * KK + j]) : -3.0e38f;

  for (int t = 1; t < TTT; ++t) {
    float e = act ? emis[((size_t)t * BB + b) * KK + j] : 0.0f;
    float best = -3.0e38f;
    int bi = 0;
#pragma unroll
    for (int i = 0; i < KK; ++i) {
      float si = __shfl(score, i, 32);
      float cand = si + tc[i];
      if (cand > best) { best = cand; bi = i; }
    }
    if (act) hist[(t - 1) * KK + j] = bi;
    float ns = best + e;
    unsigned char mt = mask[(size_t)b * TTT + t];
    score = (act && mt) ? ns : score;
  }

  score = act ? (score + end_t[j]) : -3.0e38f;
  float bs = -3.0e38f;
  int bt = 0;
#pragma unroll
  for (int i = 0; i < KK; ++i) {
    float si = __shfl(score, i, 32);
    if (si > bs) { bs = si; bt = i; }
  }
  __syncthreads();
  if (threadIdx.x == 0) {
    int tag = bt;
    out[(size_t)b * TTT + (TTT - 1)] = tag;
    for (int t = TTT - 2; t >= 0; --t) {
      unsigned char mt = mask[(size_t)b * TTT + (t + 1)];
      int prev = hist[t * KK + tag];
      tag = mt ? prev : tag;
      out[(size_t)b * TTT + t] = tag;
    }
  }
}

extern "C" void kernel_launch(void* const* d_in, const int* in_sizes, int n_in,
                              void* d_out, int out_size, void* d_ws, size_t ws_size,
                              hipStream_t stream) {
  const int*           x       = (const int*)d_in[0];
  const unsigned char* mask    = (const unsigned char*)d_in[1];
  const float*         emb     = (const float*)d_in[2];
  const float*         Wih_f   = (const float*)d_in[3];
  const float*         Whh_f   = (const float*)d_in[4];
  const float*         b_f     = (const float*)d_in[5];
  const float*         Wih_b   = (const float*)d_in[6];
  const float*         Whh_b   = (const float*)d_in[7];
  const float*         b_b     = (const float*)d_in[8];
  const float*         Wout    = (const float*)d_in[9];
  const float*         bout    = (const float*)d_in[10];
  const float*         start_t = (const float*)d_in[11];
  const float*         end_t   = (const float*)d_in[12];
  const float*         trans   = (const float*)d_in[13];
  (void)in_sizes; (void)n_in; (void)out_size; (void)ws_size;

  char* ws = (char*)d_ws;
  size_t off = 0;
  auto alloc = [&](size_t bytes) {
    void* p = ws + off;
    off += (bytes + 255) & ~(size_t)255;
    return p;
  };
  __bf16* embB  = (__bf16*)alloc((size_t)BB * TTT * EE * 2);        //   8 MB
  __bf16* xgf   = (__bf16*)alloc((size_t)BB * TTT * 4 * HH * 2);    //  64 MB (packed tiles)
  __bf16* xgb   = (__bf16*)alloc((size_t)BB * TTT * 4 * HH * 2);    //  64 MB (packed tiles)
  __bf16* feats = (__bf16*)alloc((size_t)BB * TTT * 2 * HH * 2);    //  32 MB
  float*  emis  = (float*) alloc((size_t)BB * TTT * KK * 4);        // 2.5 MB
  __bf16* pWihF = (__bf16*)alloc((size_t)EE * 4 * HH * 2);
  __bf16* pWihB = (__bf16*)alloc((size_t)EE * 4 * HH * 2);
  __bf16* pWhhF = (__bf16*)alloc((size_t)HH * 4 * HH * 2);
  __bf16* pWhhB = (__bf16*)alloc((size_t)HH * 4 * HH * 2);
  __bf16* pWout = (__bf16*)alloc((size_t)2 * HH * 32 * 2);

  // 1) embedding gather -> bf16, rows permuted to (t*B + b)
  embed_gather<<<4096, 256, 0, stream>>>(x, emb, embB, (long)BB * TTT * EE);

  // 2) pack weights into WMMA B-operand tiles
  pack_b<<<512,  256, 0, stream>>>(Wih_f, pWihF, EE,     4 * HH, 4 * HH);
  pack_b<<<512,  256, 0, stream>>>(Wih_b, pWihB, EE,     4 * HH, 4 * HH);
  pack_b<<<1024, 256, 0, stream>>>(Whh_f, pWhhF, HH,     4 * HH, 4 * HH);
  pack_b<<<1024, 256, 0, stream>>>(Whh_b, pWhhB, HH,     4 * HH, 4 * HH);
  pack_b<<<64,   256, 0, stream>>>(Wout,  pWout, 2 * HH, 32,     KK);

  // 3) xg = emb @ Wih^T + b  -> packed C-tile bf16
  {
    int waves  = ((BB * TTT) / 16) * 64;      // 131072
    int blocks = waves / 8;                   // 16384
    gemm_bpacked<1><<<blocks, 256, 0, stream>>>(embB, pWihF, b_f, xgf, BB * TTT, EE, 64, 4 * HH, 0);
    gemm_bpacked<1><<<blocks, 256, 0, stream>>>(embB, pWihB, b_b, xgb, BB * TTT, EE, 64, 4 * HH, 0);
  }

  // 4) persistent LSTM, one workgroup per direction
  lstm_kernel<<<2, 1024, 0, stream>>>(xgf, xgb, pWhhF, pWhhB, feats);

  // 5) emissions = feats @ Wout^T + bout (N padded to 32, store 20), f32
  {
    int waves  = ((BB * TTT) / 16) * 2;       // 4096
    int blocks = waves / 8;                   // 512
    gemm_bpacked<0><<<blocks, 256, 0, stream>>>(feats, pWout, bout, emis, BB * TTT, 2 * HH, 2, KK, KK);
  }

  // 6) Viterbi decode
  viterbi_kernel<<<BB, 32, 0, stream>>>(emis, mask, start_t, end_t, trans, (int*)d_out);
}
